// ScaledDotProductAttention_44633300140726
// MI455X (gfx1250) — matrix operand order repaired and checked by reading, not compiled, not run on hardware
//
#include <hip/hip_runtime.h>

// Flash-attention (causal + key-padding) BH=32, S=2048, D=64, fp32 I/O.
// Transposed-score formulation: S^T = K*Q^T so each lane owns half a softmax
// row (queries on WMMA N axis) -> per-lane softmax, zero cross-lane shuffles.
// bf16 WMMA / fp32 accumulation, 64-key double-buffered LDS tiles, software
// pipelined staging placed to fill WMMA->VALU hazard slots.

typedef __bf16 bf16;
typedef __attribute__((ext_vector_type(16))) __bf16 v16bf;
typedef __attribute__((ext_vector_type(8)))  __bf16 v8bf;
typedef __attribute__((ext_vector_type(4)))  __bf16 v4bf;
typedef __attribute__((ext_vector_type(8)))  float  v8f;

#define S_LEN   2048
#define DHEAD   64
#define BR      128                  // query rows per workgroup (16 per wave)
#define BCL     64                   // keys per tile
#define NWAVES  8
#define LOG2E   1.4426950408889634f
#define MASK_L2E  (-4294967295.0f * LOG2E)
#define SCALE_L2E (0.125f * LOG2E)   // 1/sqrt(64) * log2(e)

#define WMMA_BF16(A, B, C) \
  __builtin_amdgcn_wmma_f32_16x16x32_bf16(false, (A), false, (B), (short)0, (C), false, false)

__device__ __forceinline__ v16bf combine16(v8bf lo, v8bf hi) {
  return __builtin_shufflevector(lo, hi, 0,1,2,3,4,5,6,7,8,9,10,11,12,13,14,15);
}

// A-fragment (16 rows x 32 K, 16-bit): lane L<16 = row L, K = k0+hi*8+{0..7} and +16.
__device__ __forceinline__ v16bf load_a_frag(const bf16* base, int stride, int lane, int k0) {
  const int r  = lane & 15;
  const int hi = (lane >> 4) & 1;
  const bf16* p0 = base + r * stride + k0 + hi * 8;
  return combine16(*(const v8bf*)p0, *(const v8bf*)(p0 + 16));
}

// B-fragment (32 K x 16 cols, 16-bit): lane L<16 = col L, K = hi*8+{0..7} and +16 (matches A-fragment).
// base points at row-major storage whose row n is column n of B.
__device__ __forceinline__ v16bf load_b_frag(const bf16* base, int stride, int lane) {
  const int n  = lane & 15;
  const int hi = (lane >> 4) & 1;
  const bf16* p = base + n * stride + hi * 8;
  return combine16(*(const v8bf*)p, *(const v8bf*)(p + 16));
}

// Convert a staged K/V tile (fp32 regs) to bf16 LDS: K row-major, V transposed [d][k].
__device__ __forceinline__ void stage_tile(bf16* __restrict__ KsB, bf16* __restrict__ VtB,
                                           float* __restrict__ MsB, int tid,
                                           const float4* kr, const float4* vr, int mr) {
  #pragma unroll
  for (int i = 0; i < 4; ++i) {
    const int idx = i * 256 + tid;
    v4bf pk;
    pk[0] = (bf16)kr[i].x; pk[1] = (bf16)kr[i].y;
    pk[2] = (bf16)kr[i].z; pk[3] = (bf16)kr[i].w;
    *(v4bf*)(KsB + idx * 4) = pk;
    const int e = idx * 4, kk = e >> 6, dc = e & 63;
    VtB[(dc + 0) * BCL + kk] = (bf16)vr[i].x;
    VtB[(dc + 1) * BCL + kk] = (bf16)vr[i].y;
    VtB[(dc + 2) * BCL + kk] = (bf16)vr[i].z;
    VtB[(dc + 3) * BCL + kk] = (bf16)vr[i].w;
  }
  if (tid < BCL) MsB[tid] = (float)mr * MASK_L2E;
}

__global__ __launch_bounds__(256)
void fa_fwd_kernel(const float* __restrict__ Q, const float* __restrict__ K,
                   const float* __restrict__ V, const int* __restrict__ Mask,
                   float* __restrict__ O) {
  __shared__ __align__(16) bf16  Qs[BR * DHEAD];          // 16 KB
  __shared__ __align__(16) bf16  Ks[2][BCL * DHEAD];      // 16 KB (double buffered)
  __shared__ __align__(16) bf16  Vt[2][DHEAD * BCL];      // 16 KB ([d][k], double buffered)
  __shared__ __align__(16) float Ms[2][BCL];              // 512 B (mask * MASK * log2e)
  __shared__ __align__(16) bf16  Ps[NWAVES * 16 * BCL];   // 16 KB (per-wave P, [q][k])

  const int tid   = threadIdx.x;
  const int lane  = tid & 31;
  const int wave  = tid >> 5;
  const int hi    = (lane >> 4) & 1;
  const int q     = lane & 15;           // query column owned by this lane
  const int bh    = blockIdx.y;
  const int qbase = blockIdx.x * BR;
  const int b     = bh >> 3;             // heads = 8

  const float* Qg = Q + (size_t)(bh * S_LEN + qbase) * DHEAD;
  const float* Kg = K + (size_t)bh * S_LEN * DHEAD;
  const float* Vg = V + (size_t)bh * S_LEN * DHEAD;
  const int*   Mg = Mask + b * S_LEN;

  // ---- Stage Q tile (128x64 fp32 -> bf16 LDS) ----
  {
    const float4* q4 = (const float4*)Qg;
    #pragma unroll
    for (int i = 0; i < 8; ++i) {
      const int idx = i * 256 + tid;
      const float4 f = q4[idx];
      v4bf pk; pk[0] = (bf16)f.x; pk[1] = (bf16)f.y; pk[2] = (bf16)f.z; pk[3] = (bf16)f.w;
      *(v4bf*)(Qs + idx * 4) = pk;
    }
  }

  const int ntiles = (qbase >> 6) + (BR / BCL);   // causal upper bound

  // ---- Pipeline prologue: tile 0 into buffer 0 ----
  float4 kr[4], vr[4]; int mr;
  {
    const float4* k4 = (const float4*)Kg;
    const float4* v4 = (const float4*)Vg;
    #pragma unroll
    for (int i = 0; i < 4; ++i) { kr[i] = k4[i * 256 + tid]; vr[i] = v4[i * 256 + tid]; }
    mr = (tid < BCL) ? Mg[tid] : 0;
  }
  stage_tile(Ks[0], Vt[0], Ms[0], tid, kr, vr, mr);
  __syncthreads();

  // ---- Per-wave Q B-fragments (queries on N axis), hoisted out of key loop ----
  const bf16* qb = Qs + wave * 16 * DHEAD;
  const v16bf bq_lo = load_b_frag(qb,      DHEAD, lane);   // dims  0..31
  const v16bf bq_hi = load_b_frag(qb + 32, DHEAD, lane);   // dims 32..63

  const int rowbase = qbase + wave * 16;
  const int row     = rowbase + q;
  float m_run = -3.0e38f, l_run = 0.0f;
  v8f o0 = {}, o1 = {}, o2 = {}, o3 = {};   // O^T chunks: dim = c*16 + r + 8*hi, query = lane

  for (int jt = 0; jt < ntiles; ++jt) {
    const int cur = jt & 1;
    const bool have_next = (jt + 1) < ntiles;

    // Issue next tile's global loads early (latency hidden under QK WMMAs).
    if (have_next) {
      const float4* k4 = (const float4*)(Kg + (size_t)(jt + 1) * BCL * DHEAD);
      const float4* v4 = (const float4*)(Vg + (size_t)(jt + 1) * BCL * DHEAD);
      #pragma unroll
      for (int i = 0; i < 4; ++i) { kr[i] = k4[i * 256 + tid]; vr[i] = v4[i * 256 + tid]; }
      mr = (tid < BCL) ? Mg[(jt + 1) * BCL + tid] : 0;
    }
    if (jt + 2 < ntiles) {
      __builtin_prefetch(Kg + (size_t)(jt + 2) * BCL * DHEAD + tid * 16, 0, 1);
      __builtin_prefetch(Vg + (size_t)(jt + 2) * BCL * DHEAD + tid * 16, 0, 1);
    }

    // ---- S^T = K * Q^T : 4 x (16 keys) subtiles, C layout M=key, N=query ----
    v8f s[4];
    #pragma unroll
    for (int sub = 0; sub < 4; ++sub) {
      const bf16* kb = Ks[cur] + sub * 16 * DHEAD;
      const v16bf a0 = load_a_frag(kb, DHEAD, lane, 0);
      const v16bf a1 = load_a_frag(kb, DHEAD, lane, 32);
      v8f acc = {};
      acc = WMMA_BF16(a0, bq_lo, acc);
      acc = WMMA_BF16(a1, bq_hi, acc);
      s[sub] = acc;
    }

    // ---- Stage next tile NOW: independent cvt/ds-store work fills the
    //      WMMA->VALU hazard slots while the score WMMAs drain. ----
    if (have_next) stage_tile(Ks[cur ^ 1], Vt[cur ^ 1], Ms[cur ^ 1], tid, kr, vr, mr);

    // ---- scale + padding mask (per-key additive, staged in LDS) ----
    const int jtbase = jt * BCL;
    const float* Msp = Ms[cur];
    float x[4][8];
    #pragma unroll
    for (int sub = 0; sub < 4; ++sub)
      #pragma unroll
      for (int r = 0; r < 8; ++r)
        x[sub][r] = s[sub][r] * SCALE_L2E + Msp[sub * 16 + 8 * hi + r];

    // ---- causal mask: wave-uniform skip for tiles fully below the diagonal ----
    if (jtbase + BCL - 1 > rowbase) {
      #pragma unroll
      for (int sub = 0; sub < 4; ++sub)
        #pragma unroll
        for (int r = 0; r < 8; ++r)
          if (jtbase + sub * 16 + 8 * hi + r > row) x[sub][r] = MASK_L2E;
    }

    // ---- per-lane online softmax (lane holds keys 8*hi+r of each subtile: half the row) ----
    float mx = x[0][0];
    #pragma unroll
    for (int sub = 0; sub < 4; ++sub)
      #pragma unroll
      for (int r = 0; r < 8; ++r) mx = fmaxf(mx, x[sub][r]);
    const float mn    = fmaxf(m_run, mx);
    const float alpha = __builtin_amdgcn_exp2f(m_run - mn);
    m_run = mn;

    float ss = 0.0f;
    v8bf pk[4];
    #pragma unroll
    for (int sub = 0; sub < 4; ++sub)
      #pragma unroll
      for (int r = 0; r < 8; ++r) {
        const float p = __builtin_amdgcn_exp2f(x[sub][r] - mn);
        ss += p;
        pk[sub][r] = (bf16)p;
      }
    l_run = l_run * alpha + ss;
    o0 *= alpha; o1 *= alpha; o2 *= alpha; o3 *= alpha;

    // ---- P^T re-layout through LDS: row q as [q][key], 4x b128 stores ----
    bf16* psw = Ps + wave * 16 * BCL;
    #pragma unroll
    for (int sub = 0; sub < 4; ++sub)
      *(v8bf*)(psw + q * BCL + sub * 16 + 8 * hi) = pk[sub];

    // ---- O^T += V^T * P^T ----
    const v16bf bp0 = load_b_frag(psw,      BCL, lane);   // keys  0..31 x 16 queries
    const v16bf bp1 = load_b_frag(psw + 32, BCL, lane);   // keys 32..63 x 16 queries
    o0 = WMMA_BF16(load_a_frag(Vt[cur] + 0 * 16 * BCL, BCL, lane, 0),  bp0, o0);
    o0 = WMMA_BF16(load_a_frag(Vt[cur] + 0 * 16 * BCL, BCL, lane, 32), bp1, o0);
    o1 = WMMA_BF16(load_a_frag(Vt[cur] + 1 * 16 * BCL, BCL, lane, 0),  bp0, o1);
    o1 = WMMA_BF16(load_a_frag(Vt[cur] + 1 * 16 * BCL, BCL, lane, 32), bp1, o1);
    o2 = WMMA_BF16(load_a_frag(Vt[cur] + 2 * 16 * BCL, BCL, lane, 0),  bp0, o2);
    o2 = WMMA_BF16(load_a_frag(Vt[cur] + 2 * 16 * BCL, BCL, lane, 32), bp1, o2);
    o3 = WMMA_BF16(load_a_frag(Vt[cur] + 3 * 16 * BCL, BCL, lane, 0),  bp0, o3);
    o3 = WMMA_BF16(load_a_frag(Vt[cur] + 3 * 16 * BCL, BCL, lane, 32), bp1, o3);

    __syncthreads();   // next buffer fully staged; current buffer free
  }

  // ---- normalize (per-lane scalar) and store O: dim = c*16 + 8*hi + r ----
  const float inv = __builtin_amdgcn_rcpf(l_run);
  float* orow = O + ((size_t)bh * S_LEN + row) * DHEAD + 8 * hi;
  {
    float4 f;
    f.x=o0[0]*inv; f.y=o0[1]*inv; f.z=o0[2]*inv; f.w=o0[3]*inv; *(float4*)(orow +  0) = f;
    f.x=o0[4]*inv; f.y=o0[5]*inv; f.z=o0[6]*inv; f.w=o0[7]*inv; *(float4*)(orow +  4) = f;
    f.x=o1[0]*inv; f.y=o1[1]*inv; f.z=o1[2]*inv; f.w=o1[3]*inv; *(float4*)(orow + 16) = f;
    f.x=o1[4]*inv; f.y=o1[5]*inv; f.z=o1[6]*inv; f.w=o1[7]*inv; *(float4*)(orow + 20) = f;
    f.x=o2[0]*inv; f.y=o2[1]*inv; f.z=o2[2]*inv; f.w=o2[3]*inv; *(float4*)(orow + 32) = f;
    f.x=o2[4]*inv; f.y=o2[5]*inv; f.z=o2[6]*inv; f.w=o2[7]*inv; *(float4*)(orow + 36) = f;
    f.x=o3[0]*inv; f.y=o3[1]*inv; f.z=o3[2]*inv; f.w=o3[3]*inv; *(float4*)(orow + 48) = f;
    f.x=o3[4]*inv; f.y=o3[5]*inv; f.z=o3[6]*inv; f.w=o3[7]*inv; *(float4*)(orow + 52) = f;
  }
}

extern "C" void kernel_launch(void* const* d_in, const int* in_sizes, int n_in,
                              void* d_out, int out_size, void* d_ws, size_t ws_size,
                              hipStream_t stream) {
  (void)in_sizes; (void)n_in; (void)d_ws; (void)ws_size; (void)out_size;
  const float* Q = (const float*)d_in[0];
  const float* K = (const float*)d_in[1];
  const float* V = (const float*)d_in[2];
  const int*   M = (const int*)d_in[3];
  float* O = (float*)d_out;

  dim3 grid(S_LEN / BR, 32);   // 16 query tiles x 32 batch*heads
  fa_fwd_kernel<<<grid, 256, 0, stream>>>(Q, K, V, M, O);
}
